// Qwen3GatedDeltaNet_48962627174637
// MI455X (gfx1250) — compile-verified, matrix operand order checked by reading, not run on hardware
//
#include <hip/hip_runtime.h>
#include <hip/hip_bf16.h>
#include <cstdint>

#define L_SEQ    4096
#define HSZ      2048
#define NVH      32
#define NKH      16
#define DKD      128
#define DVD      128
#define KCONV    4
#define KEY_DIM  2048
#define VAL_DIM  4096
#define CONV_DIM 8192
#define PROJ_DIM 12352
#define PER_HEAD 772
#define EPSF     1e-6f
#define CHK      64
#define NCHK     (L_SEQ / CHK)

typedef __bf16 bf16_t;
typedef bf16_t v16bf __attribute__((ext_vector_type(16)));
typedef bf16_t v8bf  __attribute__((ext_vector_type(8)));
typedef float  v8f   __attribute__((ext_vector_type(8)));

__device__ __forceinline__ unsigned short f32_to_bf16(float f) {
  unsigned int u = __float_as_uint(f);
  u = u + 0x7FFFu + ((u >> 16) & 1u);   // round-to-nearest-even
  return (unsigned short)(u >> 16);
}

__device__ __forceinline__ float silu_f(float x) {
  return x / (1.f + __expf(-x));
}

// LDS byte offset of a generic pointer to shared memory.
__device__ __forceinline__ unsigned lds_off(const void* p) {
  return (unsigned)(uintptr_t)p;
}

// Async 32B / 16B global->LDS copies (ASYNCcnt; LDS written directly).
__device__ __forceinline__ void async_copy32(unsigned lds, const void* gsrc) {
  asm volatile("global_load_async_to_lds_b128 %0, %1, off\n\t"
               "global_load_async_to_lds_b128 %0, %1, off offset:16"
               :: "v"(lds), "v"(gsrc) : "memory");
}
__device__ __forceinline__ void wait_async0() {
  asm volatile("s_wait_asynccnt 0x0" ::: "memory");
}

// ---------------------------------------------------------------- casts
__global__ void cast_bf16_kernel(const float* __restrict__ src,
                                 unsigned short* __restrict__ dst, long n) {
  long i = (long)blockIdx.x * blockDim.x + threadIdx.x;
  long stride = (long)gridDim.x * blockDim.x;
  for (; i < n; i += stride) dst[i] = f32_to_bf16(src[i]);
}

// ======================= LDS-fed WMMA helpers (8 waves, EXEC all-1s) ========
// acc element r of v8f: m = mb + 8*(lane>>4) + r, n = nb + (lane&15)

__device__ __forceinline__ v16bf load_frag(const bf16_t* base, int row, int stride, int koff) {
  const int lane = threadIdx.x & 31;
  const bf16_t* p = base + row * stride + koff + ((lane >> 4) << 3);
  v8bf lo = *(const v8bf*)p;
  v8bf hi = *(const v8bf*)(p + 16);
  v16bf f;
  #pragma unroll
  for (int e = 0; e < 8; ++e) { f[e] = lo[e]; f[e + 8] = hi[e]; }
  return f;
}

#define WMMA_BF16(d, a, b) \
  d = __builtin_amdgcn_wmma_f32_16x16x32_bf16(false, a, false, b, (short)0, d, false, false)

// M=64, N=128: wave (wid>>2) -> 32-row slab, (wid&3) -> 32-col slab; acc[2][2]
__device__ __forceinline__ void mm_64x128(v8f acc[2][2], const bf16_t* Ab, int sA,
                                          const bf16_t* Bb, int sB, int ksteps) {
  const int lane = threadIdx.x & 31, wid = threadIdx.x >> 5;
  const int wm = (wid >> 2) * 32, wn = (wid & 3) * 32, l15 = lane & 15;
  for (int ks = 0; ks < ksteps; ++ks) {
    int ko = ks * 32;
    v16bf a0 = load_frag(Ab, wm + l15,      sA, ko);
    v16bf a1 = load_frag(Ab, wm + 16 + l15, sA, ko);
    v16bf b0 = load_frag(Bb, wn + l15,      sB, ko);
    v16bf b1 = load_frag(Bb, wn + 16 + l15, sB, ko);
    WMMA_BF16(acc[0][0], a0, b0); WMMA_BF16(acc[0][1], a0, b1);
    WMMA_BF16(acc[1][0], a1, b0); WMMA_BF16(acc[1][1], a1, b1);
  }
}

// M=64, N=64: wave (wid>>2) -> 32-row slab, (wid&3) -> 16-col slab; acc[2]
__device__ __forceinline__ void mm_64x64(v8f acc[2], const bf16_t* Ab, int sA,
                                         const bf16_t* Bb, int sB, int ksteps) {
  const int lane = threadIdx.x & 31, wid = threadIdx.x >> 5;
  const int wm = (wid >> 2) * 32, wn = (wid & 3) * 16, l15 = lane & 15;
  for (int ks = 0; ks < ksteps; ++ks) {
    int ko = ks * 32;
    v16bf a0 = load_frag(Ab, wm + l15,      sA, ko);
    v16bf a1 = load_frag(Ab, wm + 16 + l15, sA, ko);
    v16bf b0 = load_frag(Bb, wn + l15,      sB, ko);
    WMMA_BF16(acc[0], a0, b0); WMMA_BF16(acc[1], a1, b0);
  }
}

// M=128, N=128: wave (wid>>2) -> 64-row slab, (wid&3) -> 32-col slab; acc[4][2]
__device__ __forceinline__ void mm_128x128(v8f acc[4][2], const bf16_t* Ab, int sA,
                                           const bf16_t* Bb, int sB, int ksteps) {
  const int lane = threadIdx.x & 31, wid = threadIdx.x >> 5;
  const int wm = (wid >> 2) * 64, wn = (wid & 3) * 32, l15 = lane & 15;
  for (int ks = 0; ks < ksteps; ++ks) {
    int ko = ks * 32;
    v16bf b0 = load_frag(Bb, wn + l15,      sB, ko);
    v16bf b1 = load_frag(Bb, wn + 16 + l15, sB, ko);
    #pragma unroll
    for (int i = 0; i < 4; ++i) {
      v16bf a = load_frag(Ab, wm + i * 16 + l15, sA, ko);
      WMMA_BF16(acc[i][0], a, b0); WMMA_BF16(acc[i][1], a, b1);
    }
  }
}

// ---------------------------------------------------------------- WMMA GEMM
// C[M,N] = A[M,K] (bf16, row-major) x B[N,K]^T; async double-buffered tiles.
#define BM 128
#define BN 128
#define BK 32
#define LDSP 40   // padded LDS row stride in bf16 elems (80B, 16B aligned)

__global__ __launch_bounds__(256)
void gemm_bf16_wmma(const unsigned short* __restrict__ A,
                    const unsigned short* __restrict__ B,
                    float* __restrict__ C, int M, int N, int K) {
  __shared__ __align__(16) unsigned short As[2][BM * LDSP];
  __shared__ __align__(16) unsigned short Bs[2][BN * LDSP];

  const int tid  = threadIdx.x;
  const int lane = tid & 31;
  const int wid  = tid >> 5;
  const int wm   = wid >> 2;
  const int wn   = wid & 3;
  const int m_blk = blockIdx.y * BM;
  const int n_blk = blockIdx.x * BN;

  for (int i = tid; i < 2 * BM * LDSP; i += 256) {
    (&As[0][0])[i] = 0;
    (&Bs[0][0])[i] = 0;
  }

  v8f acc[4][2];
  #pragma unroll
  for (int i = 0; i < 4; ++i)
    #pragma unroll
    for (int j = 0; j < 2; ++j)
      #pragma unroll
      for (int r = 0; r < 8; ++r) acc[i][j][r] = 0.f;

  const int lr = tid >> 1;
  const int lc = (tid & 1) * 16;
  const int gm = m_blk + lr;
  const int gn = n_blk + lr;
  const unsigned ldsA0 = lds_off(&As[0][lr * LDSP + lc]);
  const unsigned ldsA1 = lds_off(&As[1][lr * LDSP + lc]);
  const unsigned ldsB0 = lds_off(&Bs[0][lr * LDSP + lc]);
  const unsigned ldsB1 = lds_off(&Bs[1][lr * LDSP + lc]);

  __syncthreads();

  if (gm < M) async_copy32(ldsA0, A + (long)gm * K + lc);
  if (gn < N) async_copy32(ldsB0, B + (long)gn * K + lc);
  wait_async0();
  __syncthreads();

  for (int kt = 0; kt < K; kt += BK) {
    const int buf = (kt >> 5) & 1;
    if (kt + BK < K) {
      unsigned la = buf ? ldsA0 : ldsA1;
      unsigned lb = buf ? ldsB0 : ldsB1;
      if (gm < M) async_copy32(la, A + (long)gm * K + kt + BK + lc);
      if (gn < N) async_copy32(lb, B + (long)gn * K + kt + BK + lc);
    }

    const int kb = (lane >> 4) * 8;
    v16bf afrag[4], bfrag[2];
    #pragma unroll
    for (int i = 0; i < 4; ++i) {
      int row = wm * 64 + i * 16 + (lane & 15);
      v8bf lo = *(const v8bf*)&As[buf][row * LDSP + kb];
      v8bf hi = *(const v8bf*)&As[buf][row * LDSP + kb + 16];
      #pragma unroll
      for (int e = 0; e < 8; ++e) { afrag[i][e] = lo[e]; afrag[i][e + 8] = hi[e]; }
    }
    #pragma unroll
    for (int j = 0; j < 2; ++j) {
      int row = wn * 32 + j * 16 + (lane & 15);
      v8bf lo = *(const v8bf*)&Bs[buf][row * LDSP + kb];
      v8bf hi = *(const v8bf*)&Bs[buf][row * LDSP + kb + 16];
      #pragma unroll
      for (int e = 0; e < 8; ++e) { bfrag[j][e] = lo[e]; bfrag[j][e + 8] = hi[e]; }
    }

    #pragma unroll
    for (int i = 0; i < 4; ++i)
      #pragma unroll
      for (int j = 0; j < 2; ++j)
        WMMA_BF16(acc[i][j], afrag[i], bfrag[j]);

    wait_async0();
    __syncthreads();
  }

  #pragma unroll
  for (int i = 0; i < 4; ++i) {
    #pragma unroll
    for (int j = 0; j < 2; ++j) {
      int n = n_blk + wn * 32 + j * 16 + (lane & 15);
      if (n >= N) continue;
      int mbase = m_blk + wm * 64 + i * 16 + (lane >> 4) * 8;
      #pragma unroll
      for (int r = 0; r < 8; ++r)
        C[(long)(mbase + r) * N + n] = acc[i][j][r];
    }
  }
}

// ---------------------------------------------------------------- conv + silu
__global__ void conv_silu_kernel(const float* __restrict__ proj,
                                 const float* __restrict__ conv_w,
                                 float* __restrict__ qc, float* __restrict__ kc,
                                 float* __restrict__ vc) {
  int c = blockIdx.x * blockDim.x + threadIdx.x;
  int t = blockIdx.y;
  int col; float* out;
  if (c < KEY_DIM) {
    int ih = c >> 7, w = c & 127;
    col = ih * PER_HEAD + w;
    out = qc + (long)t * KEY_DIM + c;
  } else if (c < 2 * KEY_DIM) {
    int cc = c - KEY_DIM, ih = cc >> 7, w = cc & 127;
    col = ih * PER_HEAD + DKD + w;
    out = kc + (long)t * KEY_DIM + cc;
  } else {
    int cc = c - 2 * KEY_DIM, vh = cc >> 7, w = cc & 127;
    col = (vh >> 1) * PER_HEAD + 2 * DKD + (vh & 1) * DVD + w;
    out = vc + (long)t * VAL_DIM + cc;
  }
  float s = 0.f;
  #pragma unroll
  for (int j = 0; j < KCONV; ++j) {
    int tt = t - (KCONV - 1) + j;
    if (tt >= 0) s += proj[(long)tt * PROJ_DIM + col] * conv_w[c * KCONV + j];
  }
  *out = silu_f(s);
}

// ---------------------------------------------------------------- q/k L2 norm
__global__ void l2norm_kernel(float* __restrict__ qc, float* __restrict__ kc) {
  __shared__ float red[128];
  int idx = blockIdx.x;
  int isK = blockIdx.y;
  int t = idx / NKH, hh = idx % NKH;
  float* base = (isK ? kc : qc) + (long)t * KEY_DIM + hh * DKD;
  int w = threadIdx.x;
  float x = base[w];
  red[w] = x * x;
  __syncthreads();
  for (int off = 64; off > 0; off >>= 1) {
    if (w < off) red[w] += red[w + off];
    __syncthreads();
  }
  float scale = rsqrtf(red[0] + EPSF);
  if (!isK) scale *= 0.08838834764831845f;   // DK^-0.5
  base[w] = x * scale;
}

// ---------------------------------------------------------------- g / beta
__global__ void gbeta_kernel(const float* __restrict__ proj,
                             const float* __restrict__ dt_bias,
                             const float* __restrict__ A_log,
                             float* __restrict__ g, float* __restrict__ beta) {
  int i = blockIdx.x * blockDim.x + threadIdx.x;
  if (i >= L_SEQ * NVH) return;
  int t = i / NVH, h = i % NVH;
  int ih = h >> 1, sub = h & 1;
  long rowoff = (long)t * PROJ_DIM + ih * PER_HEAD;
  float b = proj[rowoff + 768 + sub];
  float a = proj[rowoff + 770 + sub] + dt_bias[h];
  beta[i] = 1.f / (1.f + __expf(-b));
  float sp = (a > 20.f) ? a : log1pf(__expf(a));
  g[i] = -__expf(A_log[h]) * sp;
}

// ============================================================================
// Chunked gated delta rule (WY form), one block (8 waves) per value head.
// S_t = (I - b_t k_t^T k_t) gamma_t S_{t-1} + b_t k_t^T v_t ; chunk C=64:
//   A[i][j] = beta_i exp(G_i-G_j) (k_i.k_j)  (i>j)   [G = cumsum(g) <= 0]
//   U = (I+A)^{-1} (B V - B diag(exp G) K S_in)      [Newton/squaring, exact]
//   O = diag(exp G) Q S_in + (tril(exp(dG)) o QK^T) U
//   S = exp(G_63) S_in + (exp(G_63-G_j) k_j)^T U
// All matmuls are bf16 WMMA; state S stays f32 in LDS across chunks.
// ============================================================================
__global__ __launch_bounds__(256)
void delta_chunk_kernel(const float* __restrict__ qc, const float* __restrict__ kc,
                        const float* __restrict__ vc, const float* __restrict__ g,
                        const float* __restrict__ beta, float* __restrict__ core) {
  extern __shared__ float smem[];
  float*  S    = smem;                      // [128][128] f32 state (persistent)
  float*  U0f  = S + 128 * 128;             // [64][128] f32
  float*  KKQK = U0f + 64 * 128;            // [64][64]  f32 (KK^T then QK^T)
  float*  Gc   = KKQK + 64 * 64;            // [64] cumulative log decay
  float*  lam  = Gc + 64;                   // [64] exp(Gc)
  float*  eend = lam + 64;                  // [64] exp(Gc[63]-Gc[j])
  float*  bet  = eend + 64;                 // [64] beta
  bf16_t* SbT  = (bf16_t*)(bet + 64);       // [128][136] S^T
  bf16_t* Kr   = SbT + 128 * 136;           // [64][136] raw k
  bf16_t* Qb   = Kr + 64 * 136;             // [64][136] raw q -> scaled q
  bf16_t* KTs  = Qb + 64 * 136;             // [128][72] exp(G63-Gj) k^T
  bf16_t* U0T  = KTs + 128 * 72;            // [128][72] U0^T
  bf16_t* UT   = U0T + 128 * 72;            // [128][72] U^T
  bf16_t* Mm   = UT + 128 * 72;             // [64][72]  output mask matrix
  bf16_t* Xb   = Mm + 64 * 72;              // [64][72]  Newton X
  bf16_t* XbT  = Xb + 64 * 72;              // [64][72]  X^T
  bf16_t* Eb   = XbT + 64 * 72;             // [64][72]  Newton E
  bf16_t* EbT  = Eb + 64 * 72;              // [64][72]  E^T

  const int h = blockIdx.x, kvh = h >> 1;
  const int tid = threadIdx.x, lane = tid & 31, wid = tid >> 5;
  const int l15 = lane & 15, lhi = (lane >> 4) * 8;

  for (int i = tid; i < 128 * 128; i += 256) S[i] = 0.f;
  __syncthreads();

  for (int c = 0; c < NCHK; ++c) {
    const int t0 = c * CHK;

    // ---- scalars: cumulative decay, per-row factors
    if (tid == 0) {
      float run = 0.f;
      for (int i = 0; i < CHK; ++i) { run += g[(t0 + i) * NVH + h]; Gc[i] = run; }
    }
    __syncthreads();
    if (tid < CHK) {
      lam[tid]  = __expf(Gc[tid]);
      eend[tid] = __expf(Gc[CHK - 1] - Gc[tid]);
      bet[tid]  = beta[(t0 + tid) * NVH + h];
    }
    // ---- SbT = bf16(S^T)
    for (int i = tid; i < 128 * 128; i += 256) {
      int v = i >> 7, k = i & 127;
      SbT[v * 136 + k] = (bf16_t)S[k * 128 + v];
    }
    __syncthreads();
    // ---- stage K,Q (raw) + decay-scaled K^T
    for (int i = tid; i < CHK * 128; i += 256) {
      int j = i >> 7, d = i & 127;
      float kvv = kc[(long)(t0 + j) * KEY_DIM + kvh * DKD + d];
      float qvv = qc[(long)(t0 + j) * KEY_DIM + kvh * DKD + d];
      Kr[j * 136 + d] = (bf16_t)kvv;
      Qb[j * 136 + d] = (bf16_t)qvv;
      KTs[d * 72 + j] = (bf16_t)(kvv * eend[j]);
    }
    __syncthreads();

    // ---- U0 = B V - B diag(lam) (K S_in)
    {
      v8f acc[2][2];
      #pragma unroll
      for (int i = 0; i < 2; ++i)
        #pragma unroll
        for (int j = 0; j < 2; ++j)
          #pragma unroll
          for (int r = 0; r < 8; ++r) acc[i][j][r] = 0.f;
      mm_64x128(acc, Kr, 136, SbT, 136, 4);
      #pragma unroll
      for (int i = 0; i < 2; ++i)
        #pragma unroll
        for (int j = 0; j < 2; ++j) {
          int n = (wid & 3) * 32 + j * 16 + l15;
          #pragma unroll
          for (int r = 0; r < 8; ++r) {
            int m = (wid >> 2) * 32 + i * 16 + lhi + r;
            float vv = vc[(long)(t0 + m) * VAL_DIM + h * DVD + n];
            U0f[m * 128 + n] = bet[m] * (vv - lam[m] * acc[i][j][r]);
          }
        }
      __syncthreads();
    }
    // ---- U0T (bf16)
    for (int i = tid; i < CHK * 128; i += 256) {
      int j = i >> 7, n = i & 127;
      U0T[n * 72 + j] = (bf16_t)U0f[j * 128 + n];
    }
    // ---- KK^T
    {
      v8f acc[2];
      #pragma unroll
      for (int i = 0; i < 2; ++i)
        #pragma unroll
        for (int r = 0; r < 8; ++r) acc[i][r] = 0.f;
      mm_64x64(acc, Kr, 136, Kr, 136, 4);
      #pragma unroll
      for (int i = 0; i < 2; ++i) {
        int n = (wid & 3) * 16 + l15;
        #pragma unroll
        for (int r = 0; r < 8; ++r) {
          int m = (wid >> 2) * 32 + i * 16 + lhi + r;
          KKQK[m * 64 + n] = acc[i][r];
        }
      }
      __syncthreads();
    }
    // ---- E = -A (bf16, both layouts), X = I
    for (int i = tid; i < CHK * CHK; i += 256) {
      int ii = i >> 6, jj = i & 63;
      float a = (ii > jj) ? bet[ii] * __expf(Gc[ii] - Gc[jj]) * KKQK[ii * 64 + jj] : 0.f;
      Eb[ii * 72 + jj]  = (bf16_t)(-a);
      EbT[jj * 72 + ii] = (bf16_t)(-a);
      float xv = (ii == jj) ? 1.f : 0.f;
      Xb[ii * 72 + jj]  = (bf16_t)xv;
      XbT[jj * 72 + ii] = (bf16_t)xv;
    }
    __syncthreads();
    // ---- QK^T (overwrites KKQK)
    {
      v8f acc[2];
      #pragma unroll
      for (int i = 0; i < 2; ++i)
        #pragma unroll
        for (int r = 0; r < 8; ++r) acc[i][r] = 0.f;
      mm_64x64(acc, Qb, 136, Kr, 136, 4);
      __syncthreads();
      #pragma unroll
      for (int i = 0; i < 2; ++i) {
        int n = (wid & 3) * 16 + l15;
        #pragma unroll
        for (int r = 0; r < 8; ++r) {
          int m = (wid >> 2) * 32 + i * 16 + lhi + r;
          KKQK[m * 64 + n] = acc[i][r];
        }
      }
      __syncthreads();
    }
    // ---- Mm = tril(exp(dG) o QK^T);  Qb <- diag(lam) Qb (in place)
    for (int i = tid; i < CHK * CHK; i += 256) {
      int ii = i >> 6, jj = i & 63;
      float mval = (ii >= jj) ? __expf(Gc[ii] - Gc[jj]) * KKQK[ii * 64 + jj] : 0.f;
      Mm[ii * 72 + jj] = (bf16_t)mval;
    }
    for (int i = tid; i < CHK * 128; i += 256) {
      int j = i >> 7, d = i & 127;
      Qb[j * 136 + d] = (bf16_t)(lam[j] * (float)Qb[j * 136 + d]);
    }
    __syncthreads();

    // ---- Newton/squaring: X = (I+A)^{-1}; A^64=0 => exact after 6 iters
    #pragma unroll 1
    for (int it = 0; it < 6; ++it) {
      v8f acc[2];
      #pragma unroll
      for (int i = 0; i < 2; ++i)
        #pragma unroll
        for (int r = 0; r < 8; ++r) acc[i][r] = 0.f;
      mm_64x64(acc, Xb, 72, EbT, 72, 2);      // X*E
      __syncthreads();
      #pragma unroll
      for (int i = 0; i < 2; ++i) {
        int n = (wid & 3) * 16 + l15;
        #pragma unroll
        for (int r = 0; r < 8; ++r) {
          int m = (wid >> 2) * 32 + i * 16 + lhi + r;
          float xn = (float)Xb[m * 72 + n] + acc[i][r];
          Xb[m * 72 + n]  = (bf16_t)xn;
          XbT[n * 72 + m] = (bf16_t)xn;
        }
      }
      __syncthreads();
      if (it < 5) {
        v8f e2[2];
        #pragma unroll
        for (int i = 0; i < 2; ++i)
          #pragma unroll
          for (int r = 0; r < 8; ++r) e2[i][r] = 0.f;
        mm_64x64(e2, Eb, 72, EbT, 72, 2);     // E*E
        __syncthreads();
        #pragma unroll
        for (int i = 0; i < 2; ++i) {
          int n = (wid & 3) * 16 + l15;
          #pragma unroll
          for (int r = 0; r < 8; ++r) {
            int m = (wid >> 2) * 32 + i * 16 + lhi + r;
            Eb[m * 72 + n]  = (bf16_t)e2[i][r];
            EbT[n * 72 + m] = (bf16_t)e2[i][r];
          }
        }
        __syncthreads();
      }
    }

    // ---- U^T = (X * U0)^T
    {
      v8f acc[2][2];
      #pragma unroll
      for (int i = 0; i < 2; ++i)
        #pragma unroll
        for (int j = 0; j < 2; ++j)
          #pragma unroll
          for (int r = 0; r < 8; ++r) acc[i][j][r] = 0.f;
      mm_64x128(acc, Xb, 72, U0T, 72, 2);
      __syncthreads();
      #pragma unroll
      for (int i = 0; i < 2; ++i)
        #pragma unroll
        for (int j = 0; j < 2; ++j) {
          int n = (wid & 3) * 32 + j * 16 + l15;
          #pragma unroll
          for (int r = 0; r < 8; ++r) {
            int m = (wid >> 2) * 32 + i * 16 + lhi + r;
            UT[n * 72 + m] = (bf16_t)acc[i][j][r];
          }
        }
      __syncthreads();
    }
    // ---- O = (lam Q) S_in + Mm U  (chained accumulation) -> global
    {
      v8f acc[2][2];
      #pragma unroll
      for (int i = 0; i < 2; ++i)
        #pragma unroll
        for (int j = 0; j < 2; ++j)
          #pragma unroll
          for (int r = 0; r < 8; ++r) acc[i][j][r] = 0.f;
      mm_64x128(acc, Qb, 136, SbT, 136, 4);
      mm_64x128(acc, Mm, 72,  UT,  72,  2);
      #pragma unroll
      for (int i = 0; i < 2; ++i)
        #pragma unroll
        for (int j = 0; j < 2; ++j) {
          int n = (wid & 3) * 32 + j * 16 + l15;
          #pragma unroll
          for (int r = 0; r < 8; ++r) {
            int m = (wid >> 2) * 32 + i * 16 + lhi + r;
            core[(long)(t0 + m) * VAL_DIM + h * DVD + n] = acc[i][j][r];
          }
        }
    }
    // ---- S = exp(G63) S + KTs * U
    {
      v8f acc[4][2];
      #pragma unroll
      for (int i = 0; i < 4; ++i)
        #pragma unroll
        for (int j = 0; j < 2; ++j)
          #pragma unroll
          for (int r = 0; r < 8; ++r) acc[i][j][r] = 0.f;
      mm_128x128(acc, KTs, 72, UT, 72, 2);
      float dec = __expf(Gc[CHK - 1]);
      #pragma unroll
      for (int i = 0; i < 4; ++i)
        #pragma unroll
        for (int j = 0; j < 2; ++j) {
          int n = (wid & 3) * 32 + j * 16 + l15;
          #pragma unroll
          for (int r = 0; r < 8; ++r) {
            int m = (wid >> 2) * 64 + i * 16 + lhi + r;
            S[m * 128 + n] = dec * S[m * 128 + n] + acc[i][j][r];
          }
        }
      __syncthreads();   // next chunk reads S / reuses staging buffers
    }
  }
}

// ---------------------------------------------------------------- gate + RMSNorm -> bf16
__global__ void gate_norm_kernel(const float* __restrict__ core,
                                 const float* __restrict__ proj,
                                 const float* __restrict__ norm_w,
                                 unsigned short* __restrict__ yb) {
  __shared__ float red[128];
  int blk = blockIdx.x;
  int t = blk / NVH, h = blk % NVH;
  int w = threadIdx.x;
  int zcol = (h >> 1) * PER_HEAD + 2 * DKD + 2 * DVD + (h & 1) * DVD + w;
  float z = proj[(long)t * PROJ_DIM + zcol];
  float x = core[(long)t * VAL_DIM + h * DVD + w] * silu_f(z);
  red[w] = x * x;
  __syncthreads();
  for (int off = 64; off > 0; off >>= 1) {
    if (w < off) red[w] += red[w + off];
    __syncthreads();
  }
  float inv = rsqrtf(red[0] * (1.0f / DVD) + EPSF);
  yb[(long)t * VAL_DIM + h * DVD + w] = f32_to_bf16(x * inv * norm_w[w]);
}

// ---------------------------------------------------------------- launch
extern "C" void kernel_launch(void* const* d_in, const int* in_sizes, int n_in,
                              void* d_out, int out_size, void* d_ws, size_t ws_size,
                              hipStream_t stream) {
  const float* hs     = (const float*)d_in[0];
  const float* w_in   = (const float*)d_in[1];
  const float* conv_w = (const float*)d_in[2];
  const float* dt_b   = (const float*)d_in[3];
  const float* A_log  = (const float*)d_in[4];
  const float* norm_w = (const float*)d_in[5];
  const float* w_out  = (const float*)d_in[6];
  float* out = (float*)d_out;

  char* ws = (char*)d_ws;
  size_t off = 0;
  auto alloc = [&](size_t bytes) -> void* {
    void* p = ws + off;
    off = (off + bytes + 255) & ~(size_t)255;
    return p;
  };

  unsigned short* hsb = (unsigned short*)alloc((size_t)L_SEQ * HSZ * 2);
  unsigned short* wib = (unsigned short*)alloc((size_t)PROJ_DIM * HSZ * 2);
  unsigned short* wob = (unsigned short*)alloc((size_t)HSZ * VAL_DIM * 2);
  unsigned short* yb  = (unsigned short*)alloc((size_t)L_SEQ * VAL_DIM * 2);
  float* proj = (float*)alloc((size_t)L_SEQ * PROJ_DIM * 4);
  float* qc   = (float*)alloc((size_t)L_SEQ * KEY_DIM * 4);
  float* kc   = (float*)alloc((size_t)L_SEQ * KEY_DIM * 4);
  float* vc   = (float*)alloc((size_t)L_SEQ * VAL_DIM * 4);
  float* gbuf = (float*)alloc((size_t)L_SEQ * NVH * 4);
  float* bbuf = (float*)alloc((size_t)L_SEQ * NVH * 4);
  float* core = (float*)alloc((size_t)L_SEQ * VAL_DIM * 4);

  cast_bf16_kernel<<<512, 256, 0, stream>>>(hs, hsb, (long)L_SEQ * HSZ);
  cast_bf16_kernel<<<2048, 256, 0, stream>>>(w_in, wib, (long)PROJ_DIM * HSZ);
  cast_bf16_kernel<<<512, 256, 0, stream>>>(w_out, wob, (long)HSZ * VAL_DIM);

  dim3 g1((PROJ_DIM + BN - 1) / BN, L_SEQ / BM);
  gemm_bf16_wmma<<<g1, 256, 0, stream>>>(hsb, wib, proj, L_SEQ, PROJ_DIM, HSZ);

  dim3 g2(CONV_DIM / 256, L_SEQ);
  conv_silu_kernel<<<g2, 256, 0, stream>>>(proj, conv_w, qc, kc, vc);

  dim3 g3(L_SEQ * NKH, 2);
  l2norm_kernel<<<g3, 128, 0, stream>>>(qc, kc);

  gbeta_kernel<<<(L_SEQ * NVH + 255) / 256, 256, 0, stream>>>(proj, dt_b, A_log, gbuf, bbuf);

  // dynamic LDS: f32 region + bf16 region (~280KB of the 320KB WGP LDS)
  size_t smem_f32 = (size_t)(128 * 128 + 64 * 128 + 64 * 64 + 4 * 64) * 4;
  size_t smem_bf  = (size_t)(128 * 136 + 2 * 64 * 136 + 3 * 128 * 72 + 5 * 64 * 72) * 2;
  delta_chunk_kernel<<<NVH, 256, smem_f32 + smem_bf, stream>>>(qc, kc, vc, gbuf, bbuf, core);

  gate_norm_kernel<<<L_SEQ * NVH, 128, 0, stream>>>(core, proj, norm_w, yb);

  dim3 g4(HSZ / BN, L_SEQ / BM);
  gemm_bf16_wmma<<<g4, 256, 0, stream>>>(yb, wob, out, L_SEQ, HSZ, VAL_DIM);
}
